// EquivariantBlock_29214367547538
// MI455X (gfx1250) — matrix operand order repaired and checked by reading, not compiled, not run on hardware
//
#include <hip/hip_runtime.h>
#include <hip/hip_bf16.h>
#include <math.h>

// ---------------- problem constants ----------------
#define NN 32768
#define EE 524288
#define HH 128
#define EF 16
#define MF 273          // 2H + EF + 1
#define KP1 288         // MF padded to multiple of 32 (9 K-tiles)
#define KT1 9
#define KT2 4           // K = 128
#define KTN1 8          // K = 256
#define TILE 64         // rows (edges / nodes) per block
#define FSTR 296        // f row stride (bf16), 16B-aligned rows
#define ASTR 136        // activation row stride (bf16), 16B-aligned rows
#define NSTR 264        // node feature row stride (bf16)

// packed-weight offsets in workspace (bf16 elements); each frag = 512 elems
#define OFF_WE1 0                         // 9*8*512 = 36864
#define OFF_WC1 36864                     // 36864
#define OFF_WE2 73728                     // 4*8*512 = 16384
#define OFF_WC2 90112                     // 16384
#define OFF_WN1 106496                    // 8*8*512 = 32768
#define OFF_WN2 139264                    // 16384
#define WS_ELEMS 155648

typedef __attribute__((ext_vector_type(16))) __bf16 v16bf;
typedef __attribute__((ext_vector_type(8)))  __bf16 v8bf;
typedef __attribute__((ext_vector_type(8)))  float  v8f;

__device__ __forceinline__ float silu_f(float x) {
  return x * (1.0f / (1.0f + __expf(-x)));
}

__device__ __forceinline__ v16bf load_afrag(const __bf16* row, int kb) {
  // 16-bit A 16x32 layout: lane<16 -> K kb..kb+7 and kb+16..kb+23 (kb = kt*32)
  //                        lane>=16 -> same with kb += 8
  v8bf lo = *(const v8bf*)(row + kb);
  v8bf hi = *(const v8bf*)(row + kb + 16);
  return __builtin_shufflevector(lo, hi, 0, 1, 2, 3, 4, 5, 6, 7,
                                         8, 9, 10, 11, 12, 13, 14, 15);
}

__device__ __forceinline__ v8f wmma_bf16(v16bf a, v16bf b, v8f c) {
  return __builtin_amdgcn_wmma_f32_16x16x32_bf16(
      /*neg_a=*/false, a, /*neg_b=*/false, b,
      /*c_mod=*/(short)0, c, /*reuse_a=*/false, /*reuse_b=*/false);
}

// ---------------- kernel 0: init output + pack weights (bf16 B-fragments) ---
__device__ __forceinline__ void pack_one(const float* __restrict__ W, int Kreal,
                                         int KT, __bf16* __restrict__ dstp,
                                         long i0, long stride) {
  const long total = (long)KT * 8 * 512;
  for (long idx = i0; idx < total; idx += stride) {
    int frag = (int)(idx >> 9);
    int within = (int)(idx & 511);
    int lane = within >> 4;
    int j = within & 15;
    int kt = frag >> 3;
    int nt = frag & 7;
    int k = kt * 32 + ((j >> 3) << 4) + ((lane >> 4) << 3) + (j & 7);   // table K order (8h+s, 16+8h+s-8)
    int n = nt * 16 + (lane & 15);              // column
    float v = (k < Kreal) ? W[(long)k * HH + n] : 0.0f;
    dstp[idx] = (__bf16)v;
  }
}

__global__ void egnn_init_pack(const float* __restrict__ coords,
                               const float* __restrict__ Wc1,
                               const float* __restrict__ We1,
                               const float* __restrict__ Wc2,
                               const float* __restrict__ We2,
                               const float* __restrict__ Wn1,
                               const float* __restrict__ Wn2,
                               __bf16* __restrict__ wsb,
                               float* __restrict__ out) {
  const long i0 = (long)blockIdx.x * blockDim.x + threadIdx.x;
  const long stride = (long)gridDim.x * blockDim.x;
  for (long i = i0; i < (long)NN * HH; i += stride) out[i] = 0.0f;       // h_agg
  for (long i = i0; i < (long)NN * 3; i += stride)                       // coords seed
    out[(long)NN * HH + i] = coords[i];
  pack_one(We1, MF, KT1, wsb + OFF_WE1, i0, stride);
  pack_one(Wc1, MF, KT1, wsb + OFF_WC1, i0, stride);
  pack_one(We2, HH, KT2, wsb + OFF_WE2, i0, stride);
  pack_one(Wc2, HH, KT2, wsb + OFF_WC2, i0, stride);
  pack_one(Wn1, 2 * HH, KTN1, wsb + OFF_WN1, i0, stride);
  pack_one(Wn2, HH, KT2, wsb + OFF_WN2, i0, stride);
}

// ---------------- kernel 1: edge messages + scatter ----------------
__global__ __launch_bounds__(256) void egnn_edge_kernel(
    const int* __restrict__ src, const int* __restrict__ dst,
    const float* __restrict__ h, const float* __restrict__ coords,
    const float* __restrict__ aef,
    const float* __restrict__ b_c1, const float* __restrict__ b_c2,
    const float* __restrict__ W_c3,
    const float* __restrict__ b_e1, const float* __restrict__ b_e2,
    const float* __restrict__ W_att, const float* __restrict__ b_att,
    const __bf16* __restrict__ wsb,
    float* __restrict__ hagg, float* __restrict__ cagg) {
  __shared__ __bf16 f_lds[TILE * FSTR];
  __shared__ __bf16 e_act[TILE * ASTR];
  __shared__ __bf16 c_act[TILE * ASTR];
  __shared__ int dst_l[TILE];
  __shared__ float dif_l[TILE][3];
  __shared__ float inv_l[TILE];
  __shared__ float att_acc[TILE];
  __shared__ float scl_acc[TILE];

  const int t = threadIdx.x;
  const int lane = t & 31;
  const int wv = t >> 5;  // wave owns N-tile wv (8 tiles of 16 cols)

  // ---- phase A: build f = [h[src] | h[dst] | r2 | a | 0pad] in bf16 ----
  {
    const int e = t >> 2;
    const int sub = t & 3;
    const long ge = (long)blockIdx.x * TILE + e;
    const int s = src[ge];
    const int d = dst[ge];
    const float* hs = h + (long)s * HH + sub * 32;
    const float* hd = h + (long)d * HH + sub * 32;
    __bf16* frow = f_lds + e * FSTR;
#pragma unroll
    for (int c = 0; c < 32; c += 4) {
      float4 vs = *(const float4*)(hs + c);
      float4 vd = *(const float4*)(hd + c);
      __bf16* p0 = frow + sub * 32 + c;
      p0[0] = (__bf16)vs.x; p0[1] = (__bf16)vs.y;
      p0[2] = (__bf16)vs.z; p0[3] = (__bf16)vs.w;
      __bf16* p1 = frow + HH + sub * 32 + c;
      p1[0] = (__bf16)vd.x; p1[1] = (__bf16)vd.y;
      p1[2] = (__bf16)vd.z; p1[3] = (__bf16)vd.w;
    }
    if (sub == 0) {
      float dx = coords[(long)s * 3 + 0] - coords[(long)d * 3 + 0];
      float dy = coords[(long)s * 3 + 1] - coords[(long)d * 3 + 1];
      float dz = coords[(long)s * 3 + 2] - coords[(long)d * 3 + 2];
      float r2 = dx * dx + dy * dy + dz * dz;
      frow[2 * HH] = (__bf16)r2;
#pragma unroll
      for (int i = 0; i < EF; ++i)
        frow[2 * HH + 1 + i] = (__bf16)aef[ge * EF + i];
#pragma unroll
      for (int k = MF; k < KP1; ++k) frow[k] = (__bf16)0.0f;
      dst_l[e] = d;
      dif_l[e][0] = dx; dif_l[e][1] = dy; dif_l[e][2] = dz;
      inv_l[e] = 1.0f / (sqrtf(r2 + 1e-5f) + 1.0f);
      att_acc[e] = 0.0f;
      scl_acc[e] = 0.0f;
    }
  }
  __syncthreads();

  const int nloc = lane & 15;
  const int n = wv * 16 + nloc;
  const int mb = (lane >> 4) * 8;   // C/D row base per lane half
  const int kb0 = (lane >> 4) * 8;  // A K base per lane half

  // ---- GEMM layer 1: f[64x288] x {W_e1, W_c1}[288x128] ----
  v8f acc_e[4], acc_c[4];
#pragma unroll
  for (int mt = 0; mt < 4; ++mt) { acc_e[mt] = {}; acc_c[mt] = {}; }
  {
    const v16bf* We1 = (const v16bf*)(wsb + OFF_WE1);
    const v16bf* Wc1 = (const v16bf*)(wsb + OFF_WC1);
    for (int kt = 0; kt < KT1; ++kt) {
      v16bf be = We1[(kt * 8 + wv) * 32 + lane];
      v16bf bc = Wc1[(kt * 8 + wv) * 32 + lane];
      int kb = kt * 32 + kb0;
#pragma unroll
      for (int mt = 0; mt < 4; ++mt) {
        v16bf af = load_afrag(f_lds + (mt * 16 + nloc) * FSTR, kb);
        acc_e[mt] = wmma_bf16(af, be, acc_e[mt]);
        acc_c[mt] = wmma_bf16(af, bc, acc_c[mt]);
      }
    }
  }
  {
    float be1 = b_e1[n], bc1 = b_c1[n];
#pragma unroll
    for (int mt = 0; mt < 4; ++mt) {
#pragma unroll
      for (int j = 0; j < 8; ++j) {
        int row = mt * 16 + mb + j;
        e_act[row * ASTR + n] = (__bf16)silu_f(acc_e[mt][j] + be1);
        c_act[row * ASTR + n] = (__bf16)silu_f(acc_c[mt][j] + bc1);
      }
    }
  }
  __syncthreads();

  // ---- GEMM layer 2: act[64x128] x {W_e2, W_c2}[128x128] ----
#pragma unroll
  for (int mt = 0; mt < 4; ++mt) { acc_e[mt] = {}; acc_c[mt] = {}; }
  {
    const v16bf* We2 = (const v16bf*)(wsb + OFF_WE2);
    const v16bf* Wc2 = (const v16bf*)(wsb + OFF_WC2);
    for (int kt = 0; kt < KT2; ++kt) {
      v16bf be = We2[(kt * 8 + wv) * 32 + lane];
      v16bf bc = Wc2[(kt * 8 + wv) * 32 + lane];
      int kb = kt * 32 + kb0;
#pragma unroll
      for (int mt = 0; mt < 4; ++mt) {
        v16bf ae = load_afrag(e_act + (mt * 16 + nloc) * ASTR, kb);
        v16bf ac = load_afrag(c_act + (mt * 16 + nloc) * ASTR, kb);
        acc_e[mt] = wmma_bf16(ae, be, acc_e[mt]);
        acc_c[mt] = wmma_bf16(ac, bc, acc_c[mt]);
      }
    }
  }
  __syncthreads();  // all LDS reads of e_act/c_act complete

  // silu, stash msg_h back to e_act, partial dot-products for att / scale
  {
    float be2 = b_e2[n], bc2 = b_c2[n];
    float wat = W_att[n], wc3 = W_c3[n];
#pragma unroll
    for (int mt = 0; mt < 4; ++mt) {
#pragma unroll
      for (int j = 0; j < 8; ++j) {
        int row = mt * 16 + mb + j;
        float xe = silu_f(acc_e[mt][j] + be2);
        e_act[row * ASTR + n] = (__bf16)xe;
        atomicAdd(&att_acc[row], xe * wat);   // ds_add_f32
        float xc = silu_f(acc_c[mt][j] + bc2);
        atomicAdd(&scl_acc[row], xc * wc3);
      }
    }
  }
  __syncthreads();

  // ---- phase E: scatter-add to destination nodes ----
  const float batt = b_att[0];
  if (t < TILE) {
    const int e = t;
    float sc = scl_acc[e] * inv_l[e];
    int dd = dst_l[e];
#pragma unroll
    for (int d3 = 0; d3 < 3; ++d3)
      __hip_atomic_fetch_add(&cagg[(long)dd * 3 + d3], sc * dif_l[e][d3],
                             __ATOMIC_RELAXED, __HIP_MEMORY_SCOPE_AGENT);
  }
  {
    const int e = t >> 2;
    const int c0 = (t & 3) * 32;
    float att = 1.0f / (1.0f + __expf(-(att_acc[e] + batt)));
    int dd = dst_l[e];
    float* hp = hagg + (long)dd * HH + c0;
    const __bf16* mrow = e_act + e * ASTR + c0;
#pragma unroll
    for (int c = 0; c < 32; ++c)
      __hip_atomic_fetch_add(&hp[c], att * (float)mrow[c],
                             __ATOMIC_RELAXED, __HIP_MEMORY_SCOPE_AGENT);
  }
}

// ---------------- kernel 2: node update ----------------
__global__ __launch_bounds__(256) void egnn_node_kernel(
    const float* __restrict__ h,
    const float* __restrict__ b_n1, const float* __restrict__ b_n2,
    const __bf16* __restrict__ wsb, float* __restrict__ hio) {
  __shared__ __bf16 fn[TILE * NSTR];
  __shared__ __bf16 nact[TILE * ASTR];

  const int t = threadIdx.x;
  const int lane = t & 31;
  const int wv = t >> 5;

  {  // build [h | h_agg] in bf16; h_agg read in-place from output buffer
    const int r = t >> 2;
    const int sub = t & 3;
    const long gn = (long)blockIdx.x * TILE + r;
    const float* hr = h + gn * HH + sub * 32;
    const float* gr = hio + gn * HH + sub * 32;
    __bf16* frow = fn + r * NSTR;
#pragma unroll
    for (int c = 0; c < 32; c += 4) {
      float4 vh = *(const float4*)(hr + c);
      float4 vg = *(const float4*)(gr + c);
      __bf16* p0 = frow + sub * 32 + c;
      p0[0] = (__bf16)vh.x; p0[1] = (__bf16)vh.y;
      p0[2] = (__bf16)vh.z; p0[3] = (__bf16)vh.w;
      __bf16* p1 = frow + HH + sub * 32 + c;
      p1[0] = (__bf16)vg.x; p1[1] = (__bf16)vg.y;
      p1[2] = (__bf16)vg.z; p1[3] = (__bf16)vg.w;
    }
  }
  __syncthreads();

  const int nloc = lane & 15;
  const int n = wv * 16 + nloc;
  const int mb = (lane >> 4) * 8;
  const int kb0 = (lane >> 4) * 8;

  v8f acc[4];
#pragma unroll
  for (int mt = 0; mt < 4; ++mt) acc[mt] = {};
  {
    const v16bf* Wn1 = (const v16bf*)(wsb + OFF_WN1);
    for (int kt = 0; kt < KTN1; ++kt) {
      v16bf b = Wn1[(kt * 8 + wv) * 32 + lane];
      int kb = kt * 32 + kb0;
#pragma unroll
      for (int mt = 0; mt < 4; ++mt) {
        v16bf af = load_afrag(fn + (mt * 16 + nloc) * NSTR, kb);
        acc[mt] = wmma_bf16(af, b, acc[mt]);
      }
    }
  }
  {
    float b1 = b_n1[n];
#pragma unroll
    for (int mt = 0; mt < 4; ++mt)
#pragma unroll
      for (int j = 0; j < 8; ++j)
        nact[(mt * 16 + mb + j) * ASTR + n] = (__bf16)silu_f(acc[mt][j] + b1);
  }
  __syncthreads();

#pragma unroll
  for (int mt = 0; mt < 4; ++mt) acc[mt] = {};
  {
    const v16bf* Wn2 = (const v16bf*)(wsb + OFF_WN2);
    for (int kt = 0; kt < KT2; ++kt) {
      v16bf b = Wn2[(kt * 8 + wv) * 32 + lane];
      int kb = kt * 32 + kb0;
#pragma unroll
      for (int mt = 0; mt < 4; ++mt) {
        v16bf af = load_afrag(nact + (mt * 16 + nloc) * ASTR, kb);
        acc[mt] = wmma_bf16(af, b, acc[mt]);
      }
    }
  }
  {
    float b2 = b_n2[n];
#pragma unroll
    for (int mt = 0; mt < 4; ++mt) {
#pragma unroll
      for (int j = 0; j < 8; ++j) {
        long gm = (long)blockIdx.x * TILE + mt * 16 + mb + j;
        hio[gm * HH + n] = acc[mt][j] + b2 + h[gm * HH + n];  // residual
      }
    }
  }
}

// ---------------- host launcher ----------------
extern "C" void kernel_launch(void* const* d_in, const int* in_sizes, int n_in,
                              void* d_out, int out_size, void* d_ws,
                              size_t ws_size, hipStream_t stream) {
  const int* src = (const int*)d_in[0];
  const int* dst = (const int*)d_in[1];
  const float* h = (const float*)d_in[2];
  const float* coords = (const float*)d_in[3];
  const float* a = (const float*)d_in[4];
  const float* W_c1 = (const float*)d_in[5];
  const float* b_c1 = (const float*)d_in[6];
  const float* W_c2 = (const float*)d_in[7];
  const float* b_c2 = (const float*)d_in[8];
  const float* W_c3 = (const float*)d_in[9];
  const float* W_e1 = (const float*)d_in[10];
  const float* b_e1 = (const float*)d_in[11];
  const float* W_e2 = (const float*)d_in[12];
  const float* b_e2 = (const float*)d_in[13];
  const float* W_att = (const float*)d_in[14];
  const float* b_att = (const float*)d_in[15];
  const float* W_n1 = (const float*)d_in[16];
  const float* b_n1 = (const float*)d_in[17];
  const float* W_n2 = (const float*)d_in[18];
  const float* b_n2 = (const float*)d_in[19];

  float* out = (float*)d_out;
  float* hreg = out;                    // h_out region (used as h_agg accum)
  float* creg = out + (long)NN * HH;    // coords_out region (seeded with coords)
  __bf16* wsb = (__bf16*)d_ws;

  egnn_init_pack<<<2048, 256, 0, stream>>>(coords, W_c1, W_e1, W_c2, W_e2,
                                           W_n1, W_n2, wsb, out);
  egnn_edge_kernel<<<EE / TILE, 256, 0, stream>>>(
      src, dst, h, coords, a, b_c1, b_c2, W_c3, b_e1, b_e2, W_att, b_att, wsb,
      hreg, creg);
  egnn_node_kernel<<<NN / TILE, 256, 0, stream>>>(h, b_n1, b_n2, wsb, hreg);
}
